// GCNModel_70815420776783
// MI455X (gfx1250) — matrix operand order repaired and checked by reading, not compiled, not run on hardware
//
#include <hip/hip_runtime.h>

typedef __attribute__((ext_vector_type(16))) _Float16 v16h;
typedef __attribute__((ext_vector_type(8)))  float    v8f;

#define HD 128   // hidden / feature dim (F_IN == H == 128)

// ---------------- degree / norm ----------------
__global__ __launch_bounds__(256) void k_init_deg(float* __restrict__ deg, int n) {
  int i = blockIdx.x * 256 + threadIdx.x;
  if (i < n) deg[i] = 1.0f;                       // self-loop contributes 1
}

__global__ __launch_bounds__(256) void k_edge_deg(const int* __restrict__ dst,
                                                  float* __restrict__ deg, int e) {
  int i = blockIdx.x * 256 + threadIdx.x;
  if (i < e) atomicAdd(&deg[dst[i]], 1.0f);
}

__global__ __launch_bounds__(256) void k_rsqrt(float* __restrict__ d, int n) {
  int i = blockIdx.x * 256 + threadIdx.x;
  if (i < n) d[i] = rsqrtf(d[i]);                 // deg >= 1 always
}

// ---------------- weight convert: W[k][n] (f32) -> Wt[n][k] (f16) ----------------
__global__ __launch_bounds__(HD) void k_convW(const float* __restrict__ W,
                                              _Float16* __restrict__ Wt) {
  int k = blockIdx.x, n = threadIdx.x;
  Wt[n * HD + k] = (_Float16)W[k * HD + n];
}

// ---------------- WMMA GEMM: Y[N,128] = X[N,128] @ W[128,128] ----------------
// One wave per 16x16 output tile. grid = (N/16, 128/16), block = 32 (wave32).
// EXEC stays full (no divergent branches before the WMMAs).
__global__ __launch_bounds__(32) void k_gemm_wmma(const float* __restrict__ X,
                                                  const _Float16* __restrict__ Wt,
                                                  float* __restrict__ Y, int nrows) {
  const int tm   = blockIdx.x, tn = blockIdx.y;
  const int lane = threadIdx.x;
  const int half = lane >> 4;          // 0: lanes 0-15, 1: lanes 16-31
  const int l    = lane & 15;
  int arow = tm * 16 + l; if (arow >= nrows) arow = nrows - 1;   // clamp, keep EXEC full
  const int col  = tn * 16 + l;

  const float*    __restrict__ xrow = X  + (size_t)arow * HD;
  const _Float16* __restrict__ wcol = Wt + (size_t)col  * HD;

  v8f acc = {};
#pragma unroll
  for (int k0 = 0; k0 < HD; k0 += 32) {
    // A fragment: lane half h covers K = k0+16h..+15 (contiguous, matches B)
    const float* ap = xrow + k0 + 16 * half;
    float4 x0 = *reinterpret_cast<const float4*>(ap);
    float4 x1 = *reinterpret_cast<const float4*>(ap + 4);
    float4 x2 = *reinterpret_cast<const float4*>(ap + 8);
    float4 x3 = *reinterpret_cast<const float4*>(ap + 12);
    v16h a;
    a[0]=(_Float16)x0.x; a[1]=(_Float16)x0.y; a[2]=(_Float16)x0.z; a[3]=(_Float16)x0.w;
    a[4]=(_Float16)x1.x; a[5]=(_Float16)x1.y; a[6]=(_Float16)x1.z; a[7]=(_Float16)x1.w;
    a[8]=(_Float16)x2.x; a[9]=(_Float16)x2.y; a[10]=(_Float16)x2.z; a[11]=(_Float16)x2.w;
    a[12]=(_Float16)x3.x; a[13]=(_Float16)x3.y; a[14]=(_Float16)x3.z; a[15]=(_Float16)x3.w;
    // B fragment: 16 consecutive K for this column (pre-transposed f16) -> one 32B load
    v16h b = *reinterpret_cast<const v16h*>(wcol + k0 + 16 * half);
    acc = __builtin_amdgcn_wmma_f32_16x16x32_f16(false, a, false, b,
                                                 (short)0, acc, false, false);
  }
  // C/D layout: VGPR r -> row M = r + 8*half, column = lane&15
#pragma unroll
  for (int r = 0; r < 8; ++r) {
    int orow = tm * 16 + 8 * half + r;
    if (orow < nrows) Y[(size_t)orow * HD + tn * 16 + l] = acc[r];
  }
}

// ---------------- edge scatter: agg[dst] += h[src] * dinv[src]*dinv[dst] ----------------
// 32 threads per edge, 4 features each (float4 gather, 4 global_atomic_add_f32).
__global__ __launch_bounds__(256) void k_scatter(const int* __restrict__ src,
                                                 const int* __restrict__ dst,
                                                 const float* __restrict__ dinv,
                                                 const float* __restrict__ h,
                                                 float* __restrict__ agg, int e) {
  int t = blockIdx.x * 256 + threadIdx.x;
  int ed = t >> 5;
  if (ed >= e) return;
  int f = (t & 31) * 4;
  int s = src[ed], d = dst[ed];
  float nrm = dinv[s] * dinv[d];
  float4 hv = *reinterpret_cast<const float4*>(h + (size_t)s * HD + f);
  float* o = agg + (size_t)d * HD + f;
  atomicAdd(o + 0, hv.x * nrm);
  atomicAdd(o + 1, hv.y * nrm);
  atomicAdd(o + 2, hv.z * nrm);
  atomicAdd(o + 3, hv.w * nrm);
}

// self-loop (norm = 1/deg) + bias epilogue
__global__ __launch_bounds__(256) void k_selfloop_bias(const float* __restrict__ h,
                                                       const float* __restrict__ dinv,
                                                       const float* __restrict__ bias,
                                                       float* __restrict__ agg, int total) {
  int i = blockIdx.x * 256 + threadIdx.x;
  if (i >= total) return;
  int node = i >> 7, f = i & (HD - 1);
  float di = dinv[node];
  agg[i] += h[i] * di * di + bias[f];
}

// ---------------- batch norm (training-mode, biased var) ----------------
__global__ __launch_bounds__(HD) void k_bn_partial(const float* __restrict__ x,
                                                   float* __restrict__ sum,
                                                   float* __restrict__ sumsq,
                                                   int n, int rows_per_block) {
  int f  = threadIdx.x;
  int r0 = blockIdx.x * rows_per_block;
  int r1 = r0 + rows_per_block; if (r1 > n) r1 = n;
  float s = 0.f, ss = 0.f;
  for (int r = r0; r < r1; ++r) {
    float v = x[(size_t)r * HD + f];          // consecutive threads -> coalesced
    s += v; ss += v * v;
  }
  atomicAdd(&sum[f], s);
  atomicAdd(&sumsq[f], ss);
}

__global__ __launch_bounds__(HD) void k_bn_finalize(const float* __restrict__ sum,
                                                    const float* __restrict__ sumsq,
                                                    const float* __restrict__ gamma,
                                                    const float* __restrict__ beta,
                                                    float* __restrict__ scale,
                                                    float* __restrict__ shift, float invN) {
  int f = threadIdx.x;
  float m   = sum[f] * invN;
  float var = sumsq[f] * invN - m * m;
  float inv = rsqrtf(var + 1e-5f);
  float sc  = gamma[f] * inv;
  scale[f] = sc;
  shift[f] = beta[f] - sc * m;
}

__global__ __launch_bounds__(256) void k_bn_relu(float* __restrict__ x,
                                                 const float* __restrict__ scale,
                                                 const float* __restrict__ shift, int total) {
  int i = blockIdx.x * 256 + threadIdx.x;
  if (i >= total) return;
  int f = i & (HD - 1);
  float v = x[i] * scale[f] + shift[f];
  x[i] = v > 0.f ? v : 0.f;
}

// ---------------- global mean pool (batch is sorted -> flush on graph change) ----------------
__global__ __launch_bounds__(HD) void k_pool(const float* __restrict__ x,
                                             const int* __restrict__ batch,
                                             float* __restrict__ pooled,
                                             float* __restrict__ cnt,
                                             int n, int rows_per_block) {
  int f  = threadIdx.x;
  int r0 = blockIdx.x * rows_per_block;
  if (r0 >= n) return;
  int r1 = r0 + rows_per_block; if (r1 > n) r1 = n;
  int   g = batch[r0];
  float s = 0.f, c = 0.f;
  for (int r = r0; r < r1; ++r) {
    int gr = batch[r];
    if (gr != g) {
      atomicAdd(&pooled[g * HD + f], s);
      if (f == 0) atomicAdd(&cnt[g], c);
      s = 0.f; c = 0.f; g = gr;
    }
    s += x[(size_t)r * HD + f];
    c += 1.f;
  }
  atomicAdd(&pooled[g * HD + f], s);
  if (f == 0) atomicAdd(&cnt[g], c);
}

// ---------------- final FC: out[G,10] = (pooled/cnt) @ Wfc + bfc ----------------
__global__ __launch_bounds__(32) void k_fc(const float* __restrict__ pooled,
                                           const float* __restrict__ cnt,
                                           const float* __restrict__ Wfc,
                                           const float* __restrict__ bfc,
                                           float* __restrict__ out) {
  int g = blockIdx.x, c = threadIdx.x;
  if (c >= 10) return;
  float inv = 1.0f / fmaxf(cnt[g], 1.0f);
  float acc = bfc[c];
  for (int k = 0; k < HD; ++k) acc += pooled[g * HD + k] * inv * Wfc[k * 10 + c];
  out[g * 10 + c] = acc;
}

extern "C" void kernel_launch(void* const* d_in, const int* in_sizes, int n_in,
                              void* d_out, int out_size, void* d_ws, size_t ws_size,
                              hipStream_t stream) {
  const float* x    = (const float*)d_in[0];
  const int*   ei   = (const int*)d_in[1];
  const int*   batch= (const int*)d_in[2];
  const float* W1   = (const float*)d_in[3];
  const float* b1   = (const float*)d_in[4];
  const float* g1   = (const float*)d_in[5];
  const float* be1  = (const float*)d_in[6];
  const float* W2   = (const float*)d_in[7];
  const float* b2   = (const float*)d_in[8];
  const float* g2   = (const float*)d_in[9];
  const float* be2  = (const float*)d_in[10];
  const float* Wfc  = (const float*)d_in[11];
  const float* bfc  = (const float*)d_in[12];
  float* out = (float*)d_out;
  (void)n_in; (void)ws_size;

  const int N = in_sizes[0] / HD;
  const int E = in_sizes[1] / 2;
  const int G = out_size / 10;
  const int* esrc = ei;
  const int* edst = ei + E;

  // carve workspace (64-float aligned blocks)
  float* wsf = (float*)d_ws;
  size_t off = 0;
  auto take = [&](size_t nf) { float* p = wsf + off; off += (nf + 63) & ~(size_t)63; return p; };
  float* dinv   = take(N);
  float* bufA   = take((size_t)N * HD);   // GEMM outputs h
  float* bufB   = take((size_t)N * HD);   // aggregation / activations
  float* sum    = take(HD);
  float* sumsq  = take(HD);
  float* scale  = take(HD);
  float* shift  = take(HD);
  float* pooled = take((size_t)G * HD);
  float* cnt    = take(G);
  _Float16* W1t = (_Float16*)take((size_t)HD * HD / 2);
  _Float16* W2t = (_Float16*)take((size_t)HD * HD / 2);

  const int    nb256  = (N + 255) / 256;
  const size_t totNH  = (size_t)N * HD;
  const int    nbNH   = (int)((totNH + 255) / 256);
  const int    nbE32  = (int)(((size_t)E * 32 + 255) / 256);
  dim3 ggrid((N + 15) / 16, HD / 16);

  // degrees -> dinv
  k_init_deg<<<nb256, 256, 0, stream>>>(dinv, N);
  k_edge_deg<<<(E + 255) / 256, 256, 0, stream>>>(edst, dinv, E);
  k_rsqrt<<<nb256, 256, 0, stream>>>(dinv, N);
  // weights -> f16 transposed
  k_convW<<<HD, HD, 0, stream>>>(W1, W1t);
  k_convW<<<HD, HD, 0, stream>>>(W2, W2t);

  // ---- layer 1 ----
  k_gemm_wmma<<<ggrid, 32, 0, stream>>>(x, W1t, bufA, N);
  hipMemsetAsync(bufB, 0, totNH * sizeof(float), stream);
  k_scatter<<<nbE32, 256, 0, stream>>>(esrc, edst, dinv, bufA, bufB, E);
  k_selfloop_bias<<<nbNH, 256, 0, stream>>>(bufA, dinv, b1, bufB, (int)totNH);
  hipMemsetAsync(sum,   0, HD * sizeof(float), stream);
  hipMemsetAsync(sumsq, 0, HD * sizeof(float), stream);
  k_bn_partial<<<nb256, HD, 0, stream>>>(bufB, sum, sumsq, N, 256);
  k_bn_finalize<<<1, HD, 0, stream>>>(sum, sumsq, g1, be1, scale, shift, 1.0f / (float)N);
  k_bn_relu<<<nbNH, 256, 0, stream>>>(bufB, scale, shift, (int)totNH);

  // ---- layer 2 ----
  k_gemm_wmma<<<ggrid, 32, 0, stream>>>(bufB, W2t, bufA, N);
  hipMemsetAsync(bufB, 0, totNH * sizeof(float), stream);   // safe: GEMM2 already consumed bufB
  k_scatter<<<nbE32, 256, 0, stream>>>(esrc, edst, dinv, bufA, bufB, E);
  k_selfloop_bias<<<nbNH, 256, 0, stream>>>(bufA, dinv, b2, bufB, (int)totNH);
  hipMemsetAsync(sum,   0, HD * sizeof(float), stream);
  hipMemsetAsync(sumsq, 0, HD * sizeof(float), stream);
  k_bn_partial<<<nb256, HD, 0, stream>>>(bufB, sum, sumsq, N, 256);
  k_bn_finalize<<<1, HD, 0, stream>>>(sum, sumsq, g2, be2, scale, shift, 1.0f / (float)N);
  k_bn_relu<<<nbNH, 256, 0, stream>>>(bufB, scale, shift, (int)totNH);

  // ---- pool + fc ----
  hipMemsetAsync(pooled, 0, (size_t)G * HD * sizeof(float), stream);
  hipMemsetAsync(cnt,    0, (size_t)G * sizeof(float), stream);
  k_pool<<<nb256, HD, 0, stream>>>(bufB, batch, pooled, cnt, N, 256);
  k_fc<<<G, 32, 0, stream>>>(pooled, cnt, Wfc, bfc, out);
}
